// PointNetFeat_18889266168254
// MI455X (gfx1250) — matrix operand order repaired and checked by reading, not compiled, not run on hardware
//
#include <hip/hip_runtime.h>
#include <hip/hip_bf16.h>
#include <stdint.h>

// ---- CDNA5 WMMA / TDM operand vector types (wave32) ----
typedef __attribute__((ext_vector_type(16))) __bf16 v16bf;
typedef __attribute__((ext_vector_type(8)))  float  v8f;
typedef __attribute__((ext_vector_type(2)))  float  v2f;
typedef __attribute__((ext_vector_type(4)))  unsigned int v4u;
typedef __attribute__((ext_vector_type(8)))  int    v8i;
typedef __attribute__((ext_vector_type(4)))  int    v4i;

#define NPTS  16384
#define BATCH 4
#define EPSF  1e-5f

union BfVec { uint4 q[2]; v16bf v; unsigned short h[16]; };

// ---------------------------------------------------------------------------
// f32 -> bf16 weight conversion (once per launch; weights are tiny)
// ---------------------------------------------------------------------------
__global__ void k_cvt_bf16(const float* __restrict__ src,
                           __hip_bfloat16* __restrict__ dst, int n) {
  int i = blockIdx.x * blockDim.x + threadIdx.x;
  if (i < n) dst[i] = __float2bfloat16(src[i]);
}

// ---------------------------------------------------------------------------
// Ball query along z: one wave32 per (b, m); ballot over 32-point chunks,
// keep first-k matching indices (== reference top_k on score N-n), count all.
// ---------------------------------------------------------------------------
__global__ void k_ball_group(const float* __restrict__ pc,
                             const float* __restrict__ ctr,
                             int M, int k, float dist,
                             int* __restrict__ idx, int* __restrict__ num) {
  int wid  = (blockIdx.x * blockDim.x + threadIdx.x) >> 5;
  int lane = threadIdx.x & 31;
  if (wid >= BATCH * M) return;
  int b = wid / M, m = wid - b * M;
  float zq = ctr[(b * 3 + 2) * M + m];
  const float* z = pc + (size_t)(b * 3 + 2) * NPTS;
  int* out = idx + (size_t)wid * k;
  int cnt = 0;
  for (int base = 0; base < NPTS; base += 32) {
    float zv = z[base + lane];
    unsigned long long bal = __ballot(fabsf(zv - zq) < dist);
    if (lane == 0) {
      unsigned bits = (unsigned)bal;
      while (bits) {
        int t = __builtin_ctz(bits);
        if (cnt < k) out[cnt] = base + t;
        cnt++;
        bits &= bits - 1;
      }
    }
  }
  if (lane == 0) {
    num[wid] = cnt;
    for (int j = (cnt < k ? cnt : k); j < k; ++j) out[j] = 0;
  }
}

// ---------------------------------------------------------------------------
// Gather, point-major: X0[p][c] (c = 0..3) = {pc - center, feat}
// ---------------------------------------------------------------------------
__global__ void k_gather(const float* __restrict__ pc,
                         const float* __restrict__ feat,
                         const float* __restrict__ ctr,
                         const int* __restrict__ idx,
                         int M, int k, int P, float* __restrict__ X0) {
  int p = blockIdx.x * blockDim.x + threadIdx.x;
  if (p >= P) return;
  int mk = M * k;
  int b = p / mk;
  int r = p - b * mk;
  int m = r / k;
  int n = idx[p];
  const float* pcb = pc + (size_t)b * 3 * NPTS;
  const float* cb  = ctr + (size_t)b * 3 * M;
  float* x = X0 + (size_t)p * 4;
  x[0] = pcb[n]            - cb[m];
  x[1] = pcb[NPTS + n]     - cb[M + m];
  x[2] = pcb[2 * NPTS + n] - cb[2 * M + m];
  x[3] = feat[(size_t)b * NPTS + n];
}

// ---------------------------------------------------------------------------
// Layer-0 GEMM: Cin = 4 -> V_WMMA_F32_16X16X4_F32.
// A = X0 (16 points x 4), per-lane contiguous float2; B = W^T (4 x 16ch),
// per-lane contiguous float2 from W row. D = (point x channel) -> Y[p][o].
// ---------------------------------------------------------------------------
__global__ __launch_bounds__(256)
void k_gemm0_f32wmma(const float* __restrict__ W, const float* __restrict__ X0,
                     __hip_bfloat16* __restrict__ Y, int Cout, int P) {
  int w = threadIdx.x >> 5, lane = threadIdx.x & 31;
  int hi = lane >> 4, ln = lane & 15;
  int p0 = (blockIdx.x * 8 + w) * 16;
  int o0 = blockIdx.y * 16;
  v2f a;  // A: point row p0+ln, K pair {2*hi, 2*hi+1}
  a.x = X0[(size_t)(p0 + ln) * 4 + 2 * hi];
  a.y = X0[(size_t)(p0 + ln) * 4 + 2 * hi + 1];
  v2f bv; // B: channel col o0+ln, same K pair, contiguous in W row
  bv.x = W[(size_t)(o0 + ln) * 4 + 2 * hi];
  bv.y = W[(size_t)(o0 + ln) * 4 + 2 * hi + 1];
  v8f c = {};
  c = __builtin_amdgcn_wmma_f32_16x16x4_f32(false, a, false, bv, (short)0, c, false, false);
#pragma unroll
  for (int rr = 0; rr < 8; ++rr) {
    int p = p0 + rr + 8 * hi;
    Y[(size_t)p * Cout + o0 + ln] = __float2bfloat16(c[rr]);
  }
}

// ---------------------------------------------------------------------------
// bf16 GEMM, V_WMMA_F32_16X16X32_BF16, register-blocked 4x:
// each wave computes 16 points x 64 channels (4 WMMAs / K-step, A reused 4x).
// The block's 64 x Cin weight slab is DMA'd LDS-resident once via the Tensor
// Data Mover (tensor_load_to_lds + s_wait_tensorcnt), then B operands are two
// ds_load_b128 per tile; A operands are two contiguous 16B global loads.
// ---------------------------------------------------------------------------
__global__ __launch_bounds__(256)
void k_gemm_bf16wmma(const __hip_bfloat16* __restrict__ Wq,
                     const __hip_bfloat16* __restrict__ Xq,
                     __hip_bfloat16* __restrict__ Y,
                     int Cout, int Cin, int P,
                     const float* __restrict__ bias, int relu) {
  __shared__ alignas(32) unsigned short sW[64 * 256]; // 64 out-ch x Cin<=256
  int w = threadIdx.x >> 5, lane = threadIdx.x & 31;
  int hi = lane >> 4, ln = lane & 15;
  int p0 = (blockIdx.x * 8 + w) * 16;
  int o0 = blockIdx.y * 64;

  // --- TDM: load W[o0..o0+63][0..Cin) (bf16, row-major) into LDS ---
  if (threadIdx.x < 32) {
    unsigned long long ga =
        (unsigned long long)(uintptr_t)Wq + (unsigned long long)o0 * Cin * 2;
    unsigned ldsOff = (unsigned)(uintptr_t)(void*)sW; // generic ptr low32 = LDS addr
    v4u g0;
    g0.x = 1u;                                   // count=1, no gather
    g0.y = ldsOff;                               // lds_addr
    g0.z = (unsigned)(ga & 0xFFFFFFFFull);       // global_addr[95:64]
    g0.w = (unsigned)((ga >> 32) & 0x01FFFFFFull) | 0x80000000u; // addr hi | type=2
    v8i g1;
    g1[0] = 1 << 16;                             // wg_mask=0, data_size=2B
    g1[1] = (Cin & 0xFFFF) << 16;                // tensor_dim0 lo16
    g1[2] = ((Cin >> 16) & 0xFFFF) | ((Cout & 0xFFFF) << 16); // dim0 hi | dim1 lo
    g1[3] = ((Cout >> 16) & 0xFFFF) | ((Cin & 0xFFFF) << 16); // dim1 hi | tile_dim0
    g1[4] = 64;                                  // tile_dim1=64, tile_dim2=0
    g1[5] = Cin;                                 // tensor_dim0_stride lo32
    g1[6] = 0;
    g1[7] = 0;
    v4i gz4 = {0, 0, 0, 0};
    v8i gz8 = {0, 0, 0, 0, 0, 0, 0, 0};
    __builtin_amdgcn_tensor_load_to_lds(g0, g1, gz4, gz4, gz8, 0);
    __builtin_amdgcn_s_wait_tensorcnt(0);
  }
  __syncthreads();

  const unsigned short* X = (const unsigned short*)Xq;
  v8f acc0 = {}, acc1 = {}, acc2 = {}, acc3 = {};
  for (int kk0 = 0; kk0 < Cin; kk0 += 32) {
    // A tile: point row p0+ln; lanes 0-15 K{0..7,16..23}, 16-31 K{8..15,24..31}
    const unsigned short* xr = X + (size_t)(p0 + ln) * Cin + kk0 + hi * 8;
    BfVec av;
    av.q[0] = *(const uint4*)(xr);
    av.q[1] = *(const uint4*)(xr + 16);
    if (kk0 + 32 < Cin) __builtin_prefetch(xr + 32, 0, 1); // global_prefetch_b8
    // B tiles from LDS: channel col = j*16+ln; lanes 0-15 K{0..7,16..23}, 16-31 K{8..15,24..31}
#pragma unroll
    for (int j = 0; j < 4; ++j) {
      BfVec bv;
      const unsigned short* bp = &sW[(size_t)(j * 16 + ln) * Cin + kk0 + hi * 8];
      bv.q[0] = *(const uint4*)(bp);
      bv.q[1] = *(const uint4*)(bp + 16);
      v8f* accp = (j == 0) ? &acc0 : (j == 1) ? &acc1 : (j == 2) ? &acc2 : &acc3;
      *accp = __builtin_amdgcn_wmma_f32_16x16x32_bf16(false, av.v, false, bv.v,
                                                      (short)0, *accp, false, false);
    }
  }
#pragma unroll
  for (int j = 0; j < 4; ++j) {
    v8f* accp = (j == 0) ? &acc0 : (j == 1) ? &acc1 : (j == 2) ? &acc2 : &acc3;
    int o = o0 + j * 16 + ln;
    float bb = bias ? bias[o] : 0.0f;
#pragma unroll
    for (int rr = 0; rr < 8; ++rr) {
      int p = p0 + rr + 8 * hi;
      float v = (*accp)[rr] + bb;
      if (relu) v = fmaxf(v, 0.0f);
      Y[(size_t)p * Cout + o] = __float2bfloat16(v);
    }
  }
}

// ---------------------------------------------------------------------------
// BN stats, point-major layout, deterministic two-stage reduction:
// stage 1: 64 blocks, one thread per channel column (coalesced), partials[blk]
// stage 2: sum 64 partials per channel -> stats
// ---------------------------------------------------------------------------
__global__ void k_bn_reduce1(const __hip_bfloat16* __restrict__ Y, int C, int P,
                             float* __restrict__ partials) {
  int c = threadIdx.x; // blockDim.x == C
  float s1 = 0.f, s2 = 0.f;
  for (int p = blockIdx.x; p < P; p += gridDim.x) {
    float v = __bfloat162float(Y[(size_t)p * C + c]);
    s1 += v; s2 += v * v;
  }
  partials[(size_t)blockIdx.x * 2 * C + 2 * c]     = s1;
  partials[(size_t)blockIdx.x * 2 * C + 2 * c + 1] = s2;
}

__global__ void k_bn_reduce2(const float* __restrict__ partials, int C,
                             float* __restrict__ stats) {
  int c = blockIdx.x * blockDim.x + threadIdx.x;
  if (c >= C) return;
  float s1 = 0.f, s2 = 0.f;
  for (int blk = 0; blk < 64; ++blk) {
    s1 += partials[(size_t)blk * 2 * C + 2 * c];
    s2 += partials[(size_t)blk * 2 * C + 2 * c + 1];
  }
  stats[2 * c] = s1; stats[2 * c + 1] = s2;
}

__global__ void k_bn_finalize(const float* __restrict__ stats,
                              const float* __restrict__ g,
                              const float* __restrict__ be,
                              int C, float invP, float* __restrict__ ab) {
  int t = blockIdx.x * blockDim.x + threadIdx.x;
  if (t >= C) return;
  float mu  = stats[2 * t] * invP;
  float var = stats[2 * t + 1] * invP - mu * mu;
  float a   = g[t] * rsqrtf(var + EPSF);
  ab[2 * t]     = a;
  ab[2 * t + 1] = be[t] - mu * a;
}

__global__ void k_bn_norm_relu(const __hip_bfloat16* __restrict__ Y,
                               const float* __restrict__ ab,
                               __hip_bfloat16* __restrict__ X,
                               int C, int P, int M, int k,
                               const int* __restrict__ num) {
  size_t e = (size_t)blockIdx.x * blockDim.x + threadIdx.x; // grid == C*P/256
  int c = (int)(e % (size_t)C);
  int p = (int)(e / (size_t)C);
  float v = fmaxf(ab[2 * c] * __bfloat162float(Y[e]) + ab[2 * c + 1], 0.0f);
  if (num) {
    int mk = M * k;
    int b = p / mk;
    int m = (p - b * mk) / k;
    if (num[b * M + m] == 0) v = 0.0f;
  }
  X[e] = __float2bfloat16(v);
}

// ---------------------------------------------------------------------------
// Max over k neighbors. Threads ordered (b, m, o) with o fastest so the k
// strided reads are coalesced across the wave; output (b, o, m) per reference.
// ---------------------------------------------------------------------------
__global__ void k_max_k(const __hip_bfloat16* __restrict__ F,
                        int C, int M, int k, float* __restrict__ out) {
  int t = blockIdx.x * blockDim.x + threadIdx.x;
  if (t >= BATCH * C * M) return;
  int o = t % C;
  int r = t / C;
  int m = r % M;
  int b = r / M;
  const __hip_bfloat16* f = F + (size_t)(b * M + m) * k * C + o;
  float v = -3.0e38f;
  for (int j = 0; j < k; ++j) v = fmaxf(v, __bfloat162float(f[(size_t)j * C]));
  out[(size_t)b * C * M + (size_t)o * M + m] = v;
}

// ---------------------------------------------------------------------------
extern "C" void kernel_launch(void* const* d_in, const int* in_sizes, int n_in,
                              void* d_out, int out_size, void* d_ws, size_t ws_size,
                              hipStream_t stream) {
  (void)in_sizes; (void)n_in; (void)out_size; (void)ws_size;
  const float* pc   = (const float*)d_in[0];
  const float* feat = (const float*)d_in[1];
  const float* ctrs[4] = {(const float*)d_in[2], (const float*)d_in[3],
                          (const float*)d_in[4], (const float*)d_in[5]};

  static const int    Ms[4] = {1024, 512, 256, 128};
  static const int    Kk[4] = {32, 64, 64, 128};
  static const float  Ds[4] = {0.25f, 0.5f, 1.0f, 2.0f};
  static const int    C1[4] = {64, 64, 128, 256};
  static const int    C2[4] = {64, 64, 128, 256};
  static const int    C3[4] = {128, 128, 256, 512};
  static const int    XC[3] = {128, 128, 256};
  static const size_t outOff[4] = {0, 524288, 786432, 1048576};

  // ---- workspace carve (256B aligned slabs) ----
  char* wsBase = (char*)d_ws;
  size_t cur = 0;
  auto carve = [&](size_t bytes) -> char* {
    char* p = wsBase + cur;
    cur = (cur + bytes + 255) & ~(size_t)255;
    return p;
  };
  __hip_bfloat16* wbf1[4]; __hip_bfloat16* wbf2[4]; __hip_bfloat16* wbfx[3];
  for (int s = 0; s < 4; ++s) {
    wbf1[s] = (__hip_bfloat16*)carve((size_t)C2[s] * C1[s] * 2);
    wbf2[s] = (__hip_bfloat16*)carve((size_t)C3[s] * C2[s] * 2);
  }
  for (int s = 0; s < 3; ++s)
    wbfx[s] = (__hip_bfloat16*)carve((size_t)XC[s] * XC[s] * 2);
  int*   idxB = (int*)carve((size_t)131072 * 4);
  int*   numB = (int*)carve((size_t)4096 * 4);
  float* X0   = (float*)carve((size_t)4 * 131072 * 4);
  __hip_bfloat16* Ybuf = (__hip_bfloat16*)carve((size_t)512 * 65536 * 2); // 64MB
  __hip_bfloat16* Xbuf = (__hip_bfloat16*)carve((size_t)256 * 65536 * 2); // 32MB
  float* partials = (float*)carve((size_t)64 * 1024 * 4);
  float* stats = (float*)carve(1024 * 4);
  float* ab    = (float*)carve(1024 * 4);

  // ---- convert conv weights to bf16 once (layer0 stays f32 for x4 WMMA) ----
  for (int s = 0; s < 4; ++s) {
    const float* w1 = (const float*)d_in[6 + s * 12 + 1 * 4];
    const float* w2 = (const float*)d_in[6 + s * 12 + 2 * 4];
    int n1 = C2[s] * C1[s], n2 = C3[s] * C2[s];
    k_cvt_bf16<<<(n1 + 255) / 256, 256, 0, stream>>>(w1, wbf1[s], n1);
    k_cvt_bf16<<<(n2 + 255) / 256, 256, 0, stream>>>(w2, wbf2[s], n2);
  }
  for (int s = 0; s < 3; ++s) {
    const float* wx = (const float*)d_in[54 + s * 2];
    int nx = XC[s] * XC[s];
    k_cvt_bf16<<<(nx + 255) / 256, 256, 0, stream>>>(wx, wbfx[s], nx);
  }

  auto bn = [&](const __hip_bfloat16* Y, int C, int P, const float* g,
                const float* be, __hip_bfloat16* X, int M, int k,
                const int* num) {
    k_bn_reduce1<<<64, C, 0, stream>>>(Y, C, P, partials);
    k_bn_reduce2<<<(C + 255) / 256, 256, 0, stream>>>(partials, C, stats);
    k_bn_finalize<<<(C + 255) / 256, 256, 0, stream>>>(stats, g, be, C,
                                                       1.0f / (float)P, ab);
    k_bn_norm_relu<<<((size_t)C * P) / 256, 256, 0, stream>>>(Y, ab, X, C, P,
                                                              M, k, num);
  };

  // ---- stages ----
  for (int s = 0; s < 4; ++s) {
    int M = Ms[s], k = Kk[s], P = 4 * M * k;

    k_ball_group<<<(4 * M) / 8, 256, 0, stream>>>(pc, ctrs[s], M, k, Ds[s], idxB, numB);
    k_gather<<<P / 256, 256, 0, stream>>>(pc, feat, ctrs[s], idxB, M, k, P, X0);

    // layer 0 (Cin=4, f32 WMMA); conv bias cancels under BN -> skipped
    const float* W0 = (const float*)d_in[6 + s * 12 + 0];
    k_gemm0_f32wmma<<<dim3(P / 128, C1[s] / 16), 256, 0, stream>>>(W0, X0, Ybuf, C1[s], P);
    bn(Ybuf, C1[s], P, (const float*)d_in[6 + s * 12 + 2],
       (const float*)d_in[6 + s * 12 + 3], Xbuf, M, k, nullptr);

    // layer 1 (bf16 WMMA, TDM weight staging, 4x register-blocked)
    k_gemm_bf16wmma<<<dim3(P / 128, C2[s] / 64), 256, 0, stream>>>(
        wbf1[s], Xbuf, Ybuf, C2[s], C1[s], P, nullptr, 0);
    bn(Ybuf, C2[s], P, (const float*)d_in[6 + s * 12 + 6],
       (const float*)d_in[6 + s * 12 + 7], Xbuf, M, k, nullptr);

    // layer 2 (last MLP) — valid mask applied at normalize (before xconv)
    k_gemm_bf16wmma<<<dim3(P / 128, C3[s] / 64), 256, 0, stream>>>(
        wbf2[s], Xbuf, Ybuf, C3[s], C2[s], P, nullptr, 0);
    bn(Ybuf, C3[s], P, (const float*)d_in[6 + s * 12 + 10],
       (const float*)d_in[6 + s * 12 + 11], Xbuf, M, k, numB);

    const __hip_bfloat16* Ffinal = Xbuf;
    int Cf = C3[s];
    if (s < 3) { // xconv: bias + ReLU fused in GEMM epilogue
      k_gemm_bf16wmma<<<dim3(P / 128, XC[s] / 64), 256, 0, stream>>>(
          wbfx[s], Xbuf, Ybuf, XC[s], C3[s], P, (const float*)d_in[55 + s * 2], 1);
      Ffinal = Ybuf; Cf = XC[s];
    }
    int tot = 4 * Cf * M;
    k_max_k<<<(tot + 255) / 256, 256, 0, stream>>>(Ffinal, Cf, M, k,
                                                   (float*)d_out + outOff[s]);
  }
}
